// BinaryDense_26044681683332
// MI455X (gfx1250) — matrix operand order repaired and checked by reading, not compiled, not run on hardware
//
#include <hip/hip_runtime.h>
#include <hip/hip_bf16.h>
#include <stdint.h>

// ---------------------------------------------------------------------------
// BinaryDense: out = x @ (STD * (2*(u < sigmoid(M)) - 1))
//   x: 8192x4096 f32, M,u: 4096x4096 f32, out: 8192x4096 f32
//   STD = sqrt(2/8192) = 2^-6 exactly  -> mask = +/-2^-6, exact in bf16.
// Strategy: bf16 WMMA GEMM (v_wmma_f32_16x16x32_bf16), f32 accumulate,
// triple-buffered LDS fill via GLOBAL_LOAD_ASYNC_TO_LDS_B128 (ASYNCcnt).
// ---------------------------------------------------------------------------

typedef __attribute__((ext_vector_type(16))) __bf16 bf16x16;
typedef __attribute__((ext_vector_type(8)))  float  f32x8;

#define GEMM_M 8192
#define GEMM_N 4096
#define GEMM_K 4096
#define LDA 40            // ushorts per LDS row (80 bytes: 16B-aligned, conflict-free)
#define BUFE (128 * LDA)  // ushorts per tile buffer (10240 bytes)

__device__ __forceinline__ unsigned short f2bf(float f) {
  unsigned int x = __float_as_uint(f);
  x += 0x7FFFu + ((x >> 16) & 1u);     // round-to-nearest-even
  return (unsigned short)(x >> 16);
}

// Async global->LDS 128-bit copy (per-lane addresses), tracked by ASYNCcnt.
__device__ __forceinline__ void async_ld_b128(const unsigned short* lds,
                                              const unsigned short* g) {
  // Generic LDS pointer: low 32 bits == LDS byte offset (aperture base low = 0).
  uint32_t l = (uint32_t)(uintptr_t)lds;
  asm volatile("global_load_async_to_lds_b128 %0, %1, off"
               :: "v"(l), "v"(g) : "memory");
}

// --- Kernel 1: x (f32) -> bf16, 8 elements per thread --------------------
__global__ void __launch_bounds__(256) k_convert_x(const float* __restrict__ x,
                                                   unsigned short* __restrict__ xb) {
  const size_t i = ((size_t)blockIdx.x * 256 + threadIdx.x) * 8;
  float4 a = *(const float4*)(x + i);
  float4 b = *(const float4*)(x + i + 4);
  union { unsigned short s[8]; uint4 q; } o;
  o.s[0] = f2bf(a.x); o.s[1] = f2bf(a.y); o.s[2] = f2bf(a.z); o.s[3] = f2bf(a.w);
  o.s[4] = f2bf(b.x); o.s[5] = f2bf(b.y); o.s[6] = f2bf(b.z); o.s[7] = f2bf(b.w);
  *(uint4*)(xb + i) = o.q;
}

// --- Kernel 2: mask build + transpose: maskT[n][k] = +/-2^-6 (bf16) ------
__global__ void __launch_bounds__(256) k_mask(const float* __restrict__ M,
                                              const float* __restrict__ U,
                                              unsigned short* __restrict__ mt) {
  __shared__ unsigned short tile[32][33];
  const int n = blockIdx.x * 32 + threadIdx.x;   // column of M/u
  const int k0 = blockIdx.y * 32;                // row base of M/u
#pragma unroll
  for (int j = 0; j < 4; ++j) {
    const int k = k0 + threadIdx.y + j * 8;
    const size_t idx = (size_t)k * GEMM_N + n;
    const float p = 1.0f / (1.0f + __expf(-M[idx]));
    tile[threadIdx.y + j * 8][threadIdx.x] =
        (U[idx] < p) ? (unsigned short)0x3C80u : (unsigned short)0xBC80u; // +/-2^-6
  }
  __syncthreads();
  const int nb = blockIdx.x * 32;
#pragma unroll
  for (int j = 0; j < 4; ++j) {
    const int nr = nb + threadIdx.y + j * 8;     // output row = n
    mt[(size_t)nr * GEMM_K + k0 + threadIdx.x] = tile[threadIdx.x][threadIdx.y + j * 8];
  }
}

// --- Kernel 3: bf16 WMMA GEMM -------------------------------------------
// A: xbf   (8192 x 4096, row-major bf16)
// B: maskT (4096(N) x 4096(K), row-major bf16)
// Macro tile 128x128x32, 8 waves (2x4), wave tile 64x32 (4x2 WMMA accums).
// Triple-buffered LDS, async fills 2 tiles ahead.
__global__ void __launch_bounds__(256)
k_gemm(const unsigned short* __restrict__ A,
       const unsigned short* __restrict__ B,
       float* __restrict__ C) {
  __shared__ __align__(16) unsigned short As[3][BUFE];
  __shared__ __align__(16) unsigned short Bs[3][BUFE];   // total 60 KB

  const int tid  = threadIdx.x;
  const int lane = tid & 31;
  const int wid  = tid >> 5;
  const int wm   = wid >> 2;          // 0..1  (64-row stripe)
  const int wn   = wid & 3;           // 0..3  (32-col stripe)
  const int h    = lane >> 4;         // lane half
  const int r    = lane & 15;
  const int m0 = blockIdx.y * 128;
  const int n0 = blockIdx.x * 128;

  // Cooperative staging: 512 chunks of 16B per operand tile; 2 per thread.
  const int c0 = tid, c1 = tid + 256;
  const int rA0 = c0 >> 2, qA0 = c0 & 3;
  const int rA1 = c1 >> 2, qA1 = c1 & 3;
  const unsigned short* gA0 = A + (size_t)(m0 + rA0) * GEMM_K + qA0 * 8;
  const unsigned short* gA1 = A + (size_t)(m0 + rA1) * GEMM_K + qA1 * 8;
  const unsigned short* gB0 = B + (size_t)(n0 + rA0) * GEMM_K + qA0 * 8;
  const unsigned short* gB1 = B + (size_t)(n0 + rA1) * GEMM_K + qA1 * 8;
  const int sOff0 = rA0 * LDA + qA0 * 8;
  const int sOff1 = rA1 * LDA + qA1 * 8;

  // Fragment read offsets (ushort units) per ISA 7.12.2 layouts.
  int aOff[4], bOff[2];
#pragma unroll
  for (int i = 0; i < 4; ++i) aOff[i] = (wm * 64 + i * 16 + r) * LDA + h * 8;   // lo; hi=+16
#pragma unroll
  for (int j = 0; j < 2; ++j) bOff[j] = (wn * 32 + j * 16 + r) * LDA + h * 8;   // lo; hi=+16

  f32x8 acc[4][2];
#pragma unroll
  for (int i = 0; i < 4; ++i)
#pragma unroll
    for (int j = 0; j < 2; ++j) acc[i][j] = (f32x8){0.f,0.f,0.f,0.f,0.f,0.f,0.f,0.f};

  const int KT = GEMM_K / 32;   // 128 K-steps

  // Pipeline prologue: tiles 0 and 1 in flight (4 async ops each per wave).
  {
    async_ld_b128(&As[0][sOff0], gA0);
    async_ld_b128(&As[0][sOff1], gA1);
    async_ld_b128(&Bs[0][sOff0], gB0);
    async_ld_b128(&Bs[0][sOff1], gB1);
    async_ld_b128(&As[1][sOff0], gA0 + 32);
    async_ld_b128(&As[1][sOff1], gA1 + 32);
    async_ld_b128(&Bs[1][sOff0], gB0 + 32);
    async_ld_b128(&Bs[1][sOff1], gB1 + 32);
  }

  int cur = 0;                  // kt % 3
  for (int kt = 0; kt < KT; ++kt) {
    // Tile kt resident after this wait (tile kt+1's 4 ops may stay in flight).
    if (kt + 1 < KT) asm volatile("s_wait_asynccnt 4" ::: "memory");
    else             asm volatile("s_wait_asynccnt 0" ::: "memory");
    __syncthreads();            // publish fills; also fences last iter's reads

    if (kt + 2 < KT) {          // issue tile kt+2 into buffer (cur+2)%3
      int nb2 = cur + 2; if (nb2 >= 3) nb2 -= 3;
      const int d = (kt + 2) * 32;
      async_ld_b128(&As[nb2][sOff0], gA0 + d);
      async_ld_b128(&As[nb2][sOff1], gA1 + d);
      async_ld_b128(&Bs[nb2][sOff0], gB0 + d);
      async_ld_b128(&Bs[nb2][sOff1], gB1 + d);
    }
    if (kt + 4 < KT) {          // deep prefetch hint -> global_prefetch_b8
      __builtin_prefetch(gA0 + (kt + 4) * 32, 0, 1);
      __builtin_prefetch(gB0 + (kt + 4) * 32, 0, 1);
    }

    union Frag { bf16x16 v; uint4 q[2]; } fa[4], fb[2];
#pragma unroll
    for (int i = 0; i < 4; ++i) {
      fa[i].q[0] = *(const uint4*)&As[cur][aOff[i]];
      fa[i].q[1] = *(const uint4*)&As[cur][aOff[i] + 16];
    }
#pragma unroll
    for (int j = 0; j < 2; ++j) {
      fb[j].q[0] = *(const uint4*)&Bs[cur][bOff[j]];
      fb[j].q[1] = *(const uint4*)&Bs[cur][bOff[j] + 16];
    }

#pragma unroll
    for (int i = 0; i < 4; ++i)
#pragma unroll
      for (int j = 0; j < 2; ++j)
        acc[i][j] = __builtin_amdgcn_wmma_f32_16x16x32_bf16(
            false, fa[i].v, false, fb[j].v, (short)0, acc[i][j], false, false);

    ++cur; if (cur >= 3) cur -= 3;
  }

  // Epilogue: C layout — VGPR v: row = i*16 + h*8 + v, col = r.
#pragma unroll
  for (int i = 0; i < 4; ++i) {
    const int row0 = m0 + wm * 64 + i * 16 + h * 8;
#pragma unroll
    for (int j = 0; j < 2; ++j) {
      const int col = n0 + wn * 32 + j * 16 + r;
#pragma unroll
      for (int v = 0; v < 8; ++v)
        C[(size_t)(row0 + v) * GEMM_N + col] = acc[i][j][v];
    }
  }
}

// ---------------------------------------------------------------------------
extern "C" void kernel_launch(void* const* d_in, const int* in_sizes, int n_in,
                              void* d_out, int out_size, void* d_ws, size_t ws_size,
                              hipStream_t stream) {
  const float* x = (const float*)d_in[0];
  const float* M = (const float*)d_in[1];
  const float* u = (const float*)d_in[2];
  float* out = (float*)d_out;

  unsigned short* xbf   = (unsigned short*)d_ws;                       // 64 MB
  unsigned short* maskT = xbf + (size_t)GEMM_M * GEMM_K;               // 32 MB

  // 1) x -> bf16
  k_convert_x<<<(GEMM_M * (size_t)GEMM_K) / (256 * 8), 256, 0, stream>>>(x, xbf);
  // 2) mask build + transpose (bf16, exact +/-2^-6)
  k_mask<<<dim3(GEMM_N / 32, GEMM_K / 32), dim3(32, 8), 0, stream>>>(M, u, maskT);
  // 3) bf16 WMMA GEMM
  k_gemm<<<dim3(GEMM_N / 128, GEMM_M / 128), 256, 0, stream>>>(xbf, maskT, out);
}
